// CrossFusionBlock_22445499089236
// MI455X (gfx1250) — matrix operand order repaired and checked by reading, not compiled, not run on hardware
//
#include <hip/hip_runtime.h>
#include <hip/hip_bf16.h>
#include <math.h>

// ---------------------------------------------------------------------------
// CrossFusionBlock for MI455X (gfx1250, wave32, WMMA).
// All GEMMs via v_wmma_f32_16x16x32_bf16; WMMA-consumed intermediates
// (Q,K,V,attn-O,FFN hidden) are stored bf16 (they only ever feed WMMA),
// LayerNorm-path tensors stay fp32. Compute-bound (~137 GFLOP).
// ---------------------------------------------------------------------------

typedef __attribute__((ext_vector_type(16))) __bf16 v16bf;
typedef __attribute__((ext_vector_type(8)))  float  v8f;
typedef __attribute__((ext_vector_type(4)))  __bf16 bf16x4;
typedef __attribute__((ext_vector_type(8)))  __bf16 bf16x8;

#define LDP 8   // LDS row padding (bf16 elems): row stride 40 elems = 80 B

// ---------------------------------------------------------------------------
// GEMM: D[M,N] = act(A[M,K] * W[N,K]^T + bias)
//   AKIND 0: A fp32 row-major             (m,k) -> A[m*K + k]
//   AKIND 1: A fp32 channel-major (B,C,HW): (m,k) -> A[(m>>10)*K*1024 + k*1024 + (m&1023)]
//   AKIND 2: A bf16 row-major             (m,k) -> A[m*K + k]
//   ACT 0: bias only; ACT 1: bias + exact GELU
//   OBF 0: store fp32; OBF 1: store bf16
// Block tile 128x128, K step 32, 8 waves (2x4), wave tile 64x32 (4x2 frags).
// ---------------------------------------------------------------------------
template <int AKIND, int ACT, int OBF>
__global__ __launch_bounds__(256) void gemm_wmma_kernel(
    const void* __restrict__ Av, const float* __restrict__ W,
    const float* __restrict__ bias, void* __restrict__ Dv,
    int Ndim, int Kdim)
{
    __shared__ alignas(16) __bf16 As[128][32 + LDP];
    __shared__ alignas(16) __bf16 Bs[128][32 + LDP];   // stored [n][k]

    const float*  Af = (const float*)Av;
    const __bf16* Ah = (const __bf16*)Av;
    float*  Df = (float*)Dv;
    __bf16* Dh = (__bf16*)Dv;

    const int t    = threadIdx.x;
    const int lane = t & 31;
    const int w    = t >> 5;
    const int wm   = w >> 2;        // 0..1
    const int wn   = w & 3;         // 0..3
    const int g    = lane >> 4;     // half-wave
    const int ln   = lane & 15;
    const int bm   = blockIdx.x;    // M tile
    const int bn   = blockIdx.y;    // N tile

    v8f acc[4][2] = {};

    for (int k0 = 0; k0 < Kdim; k0 += 32) {
        // ---- stage A tile (128x32) as bf16 ----
        if (AKIND == 0) {
            // fp32, contiguous along k: float4 loads, bf16x4 LDS stores
            #pragma unroll
            for (int i = 0; i < 4; ++i) {
                int chunk = i * 256 + t;                // 1024 chunks of 4
                int c4 = (chunk & 7) * 4, r = chunk >> 3;
                const float4 v = *reinterpret_cast<const float4*>(
                    &Af[(size_t)(bm * 128 + r) * Kdim + k0 + c4]);
                bf16x4 b; b[0]=(__bf16)v.x; b[1]=(__bf16)v.y; b[2]=(__bf16)v.z; b[3]=(__bf16)v.w;
                *reinterpret_cast<bf16x4*>(&As[r][c4]) = b;
            }
        } else if (AKIND == 1) {
            // fp32 channel-major: contiguous along m, 4 strided b16 stores
            #pragma unroll
            for (int i = 0; i < 4; ++i) {
                int chunk = i * 256 + t;
                int r4 = (chunk & 31) * 4, c = chunk >> 5;
                int m = bm * 128 + r4;                  // 128-row tile stays in one b
                const float4 v = *reinterpret_cast<const float4*>(
                    &Af[(size_t)(m >> 10) * Kdim * 1024 + (size_t)(k0 + c) * 1024 + (m & 1023)]);
                As[r4 + 0][c] = (__bf16)v.x;
                As[r4 + 1][c] = (__bf16)v.y;
                As[r4 + 2][c] = (__bf16)v.z;
                As[r4 + 3][c] = (__bf16)v.w;
            }
        } else {
            // bf16 row-major: pure 16-byte copies, no conversion
            #pragma unroll
            for (int i = 0; i < 2; ++i) {
                int chunk = i * 256 + t;                // 512 chunks of 8
                int c8 = (chunk & 3) * 8, r = chunk >> 2;
                const bf16x8 v = *reinterpret_cast<const bf16x8*>(
                    &Ah[(size_t)(bm * 128 + r) * Kdim + k0 + c8]);
                *reinterpret_cast<bf16x8*>(&As[r][c8]) = v;
            }
        }
        // ---- stage B tile as Bs[n][k] (W is [N][K] fp32 row-major) ----
        #pragma unroll
        for (int i = 0; i < 4; ++i) {
            int chunk = i * 256 + t;
            int k4 = (chunk & 7) * 4, n = chunk >> 3;
            const float4 v = *reinterpret_cast<const float4*>(
                &W[(size_t)(bn * 128 + n) * Kdim + k0 + k4]);
            bf16x4 b; b[0]=(__bf16)v.x; b[1]=(__bf16)v.y; b[2]=(__bf16)v.z; b[3]=(__bf16)v.w;
            *reinterpret_cast<bf16x4*>(&Bs[n][k4]) = b;
        }
        __syncthreads();

        // ---- fragments (CDNA5 16-bit WMMA layouts) ----
        v16bf af[4], bfv[2];
        #pragma unroll
        for (int ms = 0; ms < 4; ++ms) {
            int row = wm * 64 + ms * 16 + ln;
            #pragma unroll
            for (int e = 0; e < 8; ++e)  af[ms][e] = As[row][e + 8 * g];
            #pragma unroll
            for (int e = 8; e < 16; ++e) af[ms][e] = As[row][e + 8 + 8 * g];
        }
        #pragma unroll
        for (int ns = 0; ns < 2; ++ns) {
            int col = wn * 32 + ns * 16 + ln;
            #pragma unroll
            for (int e = 0; e < 16; ++e) bfv[ns][e] = Bs[col][(e & 7) + ((e >> 3) << 4) + 8 * g];
        }
        #pragma unroll
        for (int ms = 0; ms < 4; ++ms)
            #pragma unroll
            for (int ns = 0; ns < 2; ++ns)
                acc[ms][ns] = __builtin_amdgcn_wmma_f32_16x16x32_bf16(
                    false, af[ms], false, bfv[ns], (short)0, acc[ms][ns], false, false);
        __syncthreads();
    }

    // ---- epilogue: bias (+GELU), store fp32 or bf16 row-major ----
    #pragma unroll
    for (int ms = 0; ms < 4; ++ms)
        #pragma unroll
        for (int ns = 0; ns < 2; ++ns) {
            int n = bn * 128 + wn * 32 + ns * 16 + ln;
            float bv = bias ? bias[n] : 0.f;
            #pragma unroll
            for (int r = 0; r < 8; ++r) {
                int m = bm * 128 + wm * 64 + ms * 16 + r + 8 * g;
                float v = acc[ms][ns][r] + bv;
                if (ACT == 1)
                    v = 0.5f * v * (1.f + erff(v * 0.70710678118654752f));
                if (OBF) Dh[(size_t)m * Ndim + n] = (__bf16)v;
                else     Df[(size_t)m * Ndim + n] = v;
            }
        }
}

// ---------------------------------------------------------------------------
// Flash attention on bf16 Q/K/V, bf16 O. Block = (b, head, q-group of 8
// tiles), 8 waves, each wave owns one 16-row q-tile. dh=64, heads are 64-col
// slices of C=512. kv step 32; K chunk (32x64) and transposed V chunk (64x32)
// staged in LDS with pure bf16 copies (no conversions; K b128 stores, V b16 transposed stores).
// Softmax scale folded into S after the WMMA (f32, exact for 0.125).
// ---------------------------------------------------------------------------
__global__ __launch_bounds__(256) void attn_wmma_kernel(
    const __bf16* __restrict__ Q, const __bf16* __restrict__ K,
    const __bf16* __restrict__ V, __bf16* __restrict__ O, float scale)
{
    __shared__ alignas(16) __bf16 Kl[32][64 + LDP];       // [key in chunk][dh]
    __shared__ alignas(16) __bf16 Vl[64][32 + LDP];       // [dh][key in chunk]
    __shared__ alignas(16) __bf16 pbuf[8][16][32 + LDP];  // per-wave P tile

    const int t    = threadIdx.x;
    const int lane = t & 31;
    const int w    = t >> 5;
    const int g    = lane >> 4;
    const int ln   = lane & 15;

    const int bx = blockIdx.x;         // 512 = B*HEADS*8
    const int b  = bx >> 6;
    const int h  = (bx >> 3) & 7;
    const int qg = bx & 7;
    const int q0 = (qg * 8 + w) * 16;  // this wave's q rows

    const size_t base = (size_t)b * 1024 * 512 + (size_t)h * 64;

    // staging coords: thread t covers row t>>3 (0..31), 8 cols at (t&7)*8
    const int skr = t >> 3;
    const int sc8 = (t & 7) * 8;

    // Q fragments: direct 16-byte bf16 loads
    v16bf aq[2];
    #pragma unroll
    for (int kc = 0; kc < 2; ++kc) {
        const __bf16* qp = Q + base + (size_t)(q0 + ln) * 512 + kc * 32;
        const bf16x8 lo = *reinterpret_cast<const bf16x8*>(qp + 8 * g);       // k = 8g..8g+7
        const bf16x8 hi = *reinterpret_cast<const bf16x8*>(qp + 16 + 8 * g);  // k = 16+8g..23+8g
        #pragma unroll
        for (int e = 0; e < 8; ++e) { aq[kc][e] = lo[e]; aq[kc][e + 8] = hi[e]; }
    }

    v8f o[4] = {};
    float mrow[8], lrow[8];
    #pragma unroll
    for (int r = 0; r < 8; ++r) { mrow[r] = -1e30f; lrow[r] = 0.f; }

    for (int kv = 0; kv < 1024; kv += 32) {
        // ---- cooperative staging: pure bf16 copies ----
        {
            const bf16x8 kb = *reinterpret_cast<const bf16x8*>(
                &K[base + (size_t)(kv + skr) * 512 + sc8]);
            *reinterpret_cast<bf16x8*>(&Kl[skr][sc8]) = kb;

            const bf16x8 vb = *reinterpret_cast<const bf16x8*>(
                &V[base + (size_t)(kv + skr) * 512 + sc8]);
            #pragma unroll
            for (int j = 0; j < 8; ++j) Vl[sc8 + j][skr] = vb[j];
        }
        __syncthreads();

        // ---- S = Q * K^T (16x32 as two 16x16 frags) ----
        v8f s[2] = {};
        #pragma unroll
        for (int ks = 0; ks < 2; ++ks) {
            #pragma unroll
            for (int kc = 0; kc < 2; ++kc) {
                v16bf bk;
                #pragma unroll
                for (int e = 0; e < 16; ++e)
                    bk[e] = Kl[ks * 16 + ln][kc * 32 + (e & 7) + ((e >> 3) << 4) + 8 * g];
                s[ks] = __builtin_amdgcn_wmma_f32_16x16x32_bf16(
                    false, aq[kc], false, bk, (short)0, s[ks], false, false);
            }
        }
        // fold softmax scale (exact: power of two)
        #pragma unroll
        for (int r = 0; r < 8; ++r) { s[0][r] *= scale; s[1][r] *= scale; }

        // ---- online softmax (rows live across 16 lanes of a half-wave) ----
        float pr0[8], pr1[8], alpha[8];
        #pragma unroll
        for (int r = 0; r < 8; ++r) {
            float mx = fmaxf(s[0][r], s[1][r]);
            #pragma unroll
            for (int off = 1; off < 16; off <<= 1)
                mx = fmaxf(mx, __shfl_xor(mx, off, 32));
            float mn = fmaxf(mrow[r], mx);
            float al = __expf(mrow[r] - mn);
            mrow[r] = mn;
            float p0 = __expf(s[0][r] - mn);
            float p1 = __expf(s[1][r] - mn);
            float rs = p0 + p1;
            #pragma unroll
            for (int off = 1; off < 16; off <<= 1)
                rs += __shfl_xor(rs, off, 32);
            lrow[r] = lrow[r] * al + rs;
            alpha[r] = al; pr0[r] = p0; pr1[r] = p1;
        }
        #pragma unroll
        for (int ns = 0; ns < 4; ++ns)
            #pragma unroll
            for (int r = 0; r < 8; ++r) o[ns][r] *= alpha[r];

        // ---- P: D-layout -> A-layout via per-wave LDS tile ----
        #pragma unroll
        for (int r = 0; r < 8; ++r) {
            pbuf[w][r + 8 * g][ln]      = (__bf16)pr0[r];
            pbuf[w][r + 8 * g][16 + ln] = (__bf16)pr1[r];
        }
        __syncthreads();   // pbuf visibility; all waves convergent
        v16bf pa;
        #pragma unroll
        for (int e = 0; e < 8; ++e)  pa[e] = pbuf[w][ln][e + 8 * g];
        #pragma unroll
        for (int e = 8; e < 16; ++e) pa[e] = pbuf[w][ln][e + 8 + 8 * g];

        // ---- O += P * V ----
        #pragma unroll
        for (int ns = 0; ns < 4; ++ns) {
            v16bf bv;
            #pragma unroll
            for (int e = 0; e < 16; ++e)
                bv[e] = Vl[ns * 16 + ln][(e & 7) + ((e >> 3) << 4) + 8 * g];
            o[ns] = __builtin_amdgcn_wmma_f32_16x16x32_bf16(
                false, pa, false, bv, (short)0, o[ns], false, false);
        }
        __syncthreads();   // done with Kl/Vl/pbuf before next staging
    }

    // ---- finalize: O / l, store bf16 ----
    #pragma unroll
    for (int r = 0; r < 8; ++r) {
        float inv = 1.f / lrow[r];
        int m = q0 + r + 8 * g;
        #pragma unroll
        for (int ns = 0; ns < 4; ++ns)
            O[base + (size_t)m * 512 + ns * 16 + ln] = (__bf16)(o[ns][r] * inv);
    }
}

// ---------------------------------------------------------------------------
// Residual add + channel LayerNorm over C=512 at each (b,hw) position (fp32).
//   Ain : row-major (m, C) addend
//   Bin : second addend; b_mode 0 = row-major, 1 = (B,C,HW) channel-major
//   out : out_mode 0 = row-major, 1 = (B,C,HW) channel-major
// ---------------------------------------------------------------------------
__global__ __launch_bounds__(256) void ln_res_kernel(
    const float* __restrict__ Ain, const float* __restrict__ Bin, int b_mode,
    const float* __restrict__ wgt, const float* __restrict__ bia,
    float* __restrict__ out, int out_mode)
{
    __shared__ float red[256];
    const int m  = blockIdx.x;       // 0..8191
    const int t  = threadIdx.x;
    const int bb = m >> 10, hw = m & 1023;

    float v[2];
    float s = 0.f;
    #pragma unroll
    for (int i = 0; i < 2; ++i) {
        int c = t + i * 256;
        float a = Ain[(size_t)m * 512 + c];
        float x = (b_mode == 0) ? Bin[(size_t)m * 512 + c]
                                : Bin[(size_t)bb * 512 * 1024 + (size_t)c * 1024 + hw];
        v[i] = a + x; s += v[i];
    }
    red[t] = s; __syncthreads();
    for (int o2 = 128; o2 > 0; o2 >>= 1) {
        if (t < o2) red[t] += red[t + o2];
        __syncthreads();
    }
    float mu = red[0] * (1.f / 512.f);
    __syncthreads();
    float sq = 0.f;
    #pragma unroll
    for (int i = 0; i < 2; ++i) { float d = v[i] - mu; sq += d * d; }
    red[t] = sq; __syncthreads();
    for (int o2 = 128; o2 > 0; o2 >>= 1) {
        if (t < o2) red[t] += red[t + o2];
        __syncthreads();
    }
    float rstd = rsqrtf(red[0] * (1.f / 512.f) + 1e-6f);
    #pragma unroll
    for (int i = 0; i < 2; ++i) {
        int c = t + i * 256;
        float y = (v[i] - mu) * rstd * wgt[c] + bia[c];
        if (out_mode == 0) out[(size_t)m * 512 + c] = y;
        else out[(size_t)bb * 512 * 1024 + (size_t)c * 1024 + hw] = y;
    }
}

// ---------------------------------------------------------------------------
// Host orchestration
// ---------------------------------------------------------------------------
extern "C" void kernel_launch(void* const* d_in, const int* in_sizes, int n_in,
                              void* d_out, int out_size, void* d_ws, size_t ws_size,
                              hipStream_t stream) {
    (void)in_sizes; (void)n_in; (void)out_size; (void)ws_size;

    const float* xs = (const float*)d_in[0];   // spatial_feat (B,C,H,W)
    const float* xf = (const float*)d_in[1];   // freq_feat

    const size_t MC = (size_t)8192 * 512;      // elements of one (M,C) buffer
    // bf16 region: Q,K,V,O (MC each) + H (M x 2048)
    __bf16* Qb = (__bf16*)d_ws;
    __bf16* Kb = Qb + MC;
    __bf16* Vb = Qb + 2 * MC;
    __bf16* Ob = Qb + 3 * MC;
    __bf16* Hb = Qb + 4 * MC;                  // 8192 x 2048 bf16
    // fp32 region after bf16 region (4*MC + 4*MC halves = 8*MC bf16 = 4*MC floats)
    float*  ENH = (float*)(Qb + 8 * MC);
    float*  Y1  = ENH + MC;
    float*  F2  = ENH + 2 * MC;

    const dim3 blk(256);
    const dim3 gP(8192 / 128, 512 / 128);      // 64 x 4   (N=512 GEMMs)
    const dim3 gH(8192 / 128, 2048 / 128);     // 64 x 16  (N=2048 GEMM)

    for (int sidx = 0; sidx < 2; ++sidx) {
        const float* qmap  = (sidx == 0) ? xs : xf;
        const float* kvmap = (sidx == 0) ? xf : xs;
        const float* Wq  = (const float*)d_in[sidx == 0 ? 2 : 7];
        const float* Wk  = (const float*)d_in[sidx == 0 ? 3 : 8];
        const float* Wv  = (const float*)d_in[sidx == 0 ? 4 : 9];
        const float* Wo  = (const float*)d_in[sidx == 0 ? 5 : 10];
        const float* bo  = (const float*)d_in[sidx == 0 ? 6 : 11];
        const float* n1w = (const float*)d_in[sidx == 0 ? 12 : 16];
        const float* n1b = (const float*)d_in[sidx == 0 ? 13 : 17];
        const float* n2w = (const float*)d_in[sidx == 0 ? 14 : 18];
        const float* n2b = (const float*)d_in[sidx == 0 ? 15 : 19];
        const float* W1  = (const float*)d_in[sidx == 0 ? 20 : 24];
        const float* b1  = (const float*)d_in[sidx == 0 ? 21 : 25];
        const float* W2  = (const float*)d_in[sidx == 0 ? 22 : 26];
        const float* b2  = (const float*)d_in[sidx == 0 ? 23 : 27];

        // Q/K/V projections: fp32 channel-major A -> bf16 outputs
        gemm_wmma_kernel<1,0,1><<<gP, blk, 0, stream>>>(qmap,  Wq, nullptr, Qb, 512, 512);
        gemm_wmma_kernel<1,0,1><<<gP, blk, 0, stream>>>(kvmap, Wk, nullptr, Kb, 512, 512);
        gemm_wmma_kernel<1,0,1><<<gP, blk, 0, stream>>>(kvmap, Wv, nullptr, Vb, 512, 512);

        // Flash attention (scale = dh^-0.5 = 0.125), bf16 in/out
        attn_wmma_kernel<<<512, blk, 0, stream>>>(Qb, Kb, Vb, Ob, 0.125f);

        // Output projection + bias: bf16 A -> fp32 out
        gemm_wmma_kernel<2,0,0><<<gP, blk, 0, stream>>>(Ob, Wo, bo, ENH, 512, 512);

        // LN1: LN_c(x + enh) -> Y1 (fp32 row-major)
        ln_res_kernel<<<8192, blk, 0, stream>>>(ENH, qmap, 1, n1w, n1b, Y1, 0);

        // FFN: H = GELU(Y1 W1^T + b1) bf16;  F2 = H W2^T + b2 fp32
        gemm_wmma_kernel<0,1,1><<<gH, blk, 0, stream>>>(Y1, W1, b1, Hb, 2048, 512);
        gemm_wmma_kernel<2,0,0><<<gP, blk, 0, stream>>>(Hb, W2, b2, F2, 512, 2048);

        // LN2: LN_c(Y1 + F2) -> d_out (channel-major (B,C,H,W))
        ln_res_kernel<<<8192, blk, 0, stream>>>(F2, Y1, 0, n2w, n2b,
                                                (float*)d_out + sidx * MC, 1);
    }
}
